// LSTM_Calib_9972914061687
// MI455X (gfx1250) — matrix operand order repaired and checked by reading, not compiled, not run on hardware
//
#include <hip/hip_runtime.h>

// MI455X / gfx1250 fused 2-layer LSTM + collapsed linear head.
// One wave32 handles 16 batch samples. Per timestep per layer: 4x
// v_wmma_f32_16x16x32_f16 computes G^T = W * feat^T (bias folded into A via a
// constant-1 K-column; layer-0 scalar input folded into K=16). State updates
// are per-lane VALU (v_tanh_f32 / v_exp_f32 TRANS ops co-execute with WMMA);
// cross-half exchange via ds_swizzle SWAPX16.

typedef __attribute__((ext_vector_type(16))) _Float16 v16h;
typedef __attribute__((ext_vector_type(8)))  float    v8f;
typedef __attribute__((ext_vector_type(8)))  int      vi8;
typedef __attribute__((ext_vector_type(2)))  __fp16   fp16x2;

__device__ __forceinline__ float lane_xor16(float v) {
    // group-of-32 swizzle: and=0x1f, or=0, xor=0x10 -> SWAPX16
    return __int_as_float(__builtin_amdgcn_ds_swizzle(__float_as_int(v), 0x401f));
}

// Fast activations on the TRANS pipe. gfx1250 has V_TANH_F32; guard the
// builtin name and fall back to v_exp + v_rcp (never precise division).
#if __has_builtin(__builtin_amdgcn_tanhf)
__device__ __forceinline__ float ftanh(float x) { return __builtin_amdgcn_tanhf(x); }
__device__ __forceinline__ float fsig(float x)  { return __builtin_fmaf(0.5f, __builtin_amdgcn_tanhf(0.5f * x), 0.5f); }
#elif __has_builtin(__builtin_amdgcn_tanh_f32)
__device__ __forceinline__ float ftanh(float x) { return __builtin_amdgcn_tanh_f32(x); }
__device__ __forceinline__ float fsig(float x)  { return __builtin_fmaf(0.5f, __builtin_amdgcn_tanh_f32(0.5f * x), 0.5f); }
#else
__device__ __forceinline__ float ftanh(float x) {
    return 1.0f - 2.0f * __builtin_amdgcn_rcpf(1.0f + __expf(2.0f * x));
}
__device__ __forceinline__ float fsig(float x) {
    return __builtin_amdgcn_rcpf(1.0f + __expf(-x));
}
#endif

__device__ __forceinline__ int pk2(float a, float b) {
    fp16x2 p = __builtin_amdgcn_cvt_pkrtz(a, b);
    return __builtin_bit_cast(int, p);
}

// A-element generators. K is always a compile-time constant at every call
// site (loops fully unrolled, both lane-halves computed explicitly), so the
// K comparisons fold away and loads are unconditional (row pre-clamped).
// Layer-0: K<15 -> W_hh0[row][K] ; K==15 -> bias ; K==16 -> W_ih0[row] ; else 0
__device__ __forceinline__ float a0e(int K, int row,
        const float* __restrict__ Whh0, const float* __restrict__ Wih0,
        const float* __restrict__ bi0, const float* __restrict__ bh0) {
    if (K < 15)  return Whh0[row * 15 + K];
    if (K == 15) return bi0[row] + bh0[row];
    if (K == 16) return Wih0[row];
    return 0.0f;
}
// Layer-1: K<15 -> W_ih1[row][K] ; K==15 -> bias ; 16<=K<31 -> W_hh1[row][K-16]
__device__ __forceinline__ float a1e(int K, int row,
        const float* __restrict__ Wih1, const float* __restrict__ Whh1,
        const float* __restrict__ bi1, const float* __restrict__ bh1) {
    if (K < 15)  return Wih1[row * 15 + K];
    if (K == 15) return bi1[row] + bh1[row];
    if (K >= 16 && K < 31) return Whh1[row * 15 + (K - 16)];
    return 0.0f;
}

// Collapse the purely-linear head: w_eff = W1^T W2^T W3^T (15 floats),
// b_eff = (b1 W2^T + b2) W3^T + b3. Stored in d_ws[0..15].
__global__ void head_collapse_kernel(const float* __restrict__ W1,
                                     const float* __restrict__ b1,
                                     const float* __restrict__ W2,
                                     const float* __restrict__ b2,
                                     const float* __restrict__ W3,
                                     const float* __restrict__ b3,
                                     float* __restrict__ head) {
    int k = threadIdx.x;
    if (k < 15) {
        float acc = 0.0f;
        for (int i = 0; i < 20; ++i) {
            float t = 0.0f;
            for (int j = 0; j < 60; ++j) t += W2[i * 60 + j] * W1[j * 15 + k];
            acc += W3[i] * t;
        }
        head[k] = acc;
    } else if (k == 15) {
        float acc = b3[0];
        for (int i = 0; i < 20; ++i) {
            float t = b2[i];
            for (int j = 0; j < 60; ++j) t += W2[i * 60 + j] * b1[j];
            acc += W3[i] * t;
        }
        head[15] = acc;
    }
}

__global__ __launch_bounds__(256, 1)
void lstm_fused_wmma(const float* __restrict__ x,
                     const float* __restrict__ Wih0, const float* __restrict__ Whh0,
                     const float* __restrict__ bi0,  const float* __restrict__ bh0,
                     const float* __restrict__ Wih1, const float* __restrict__ Whh1,
                     const float* __restrict__ bi1,  const float* __restrict__ bh1,
                     const float* __restrict__ head,
                     float* __restrict__ out, int B) {
    const int lane = (int)(threadIdx.x & 31u);
    const int m    = lane & 15;     // A: row M ; B/D: batch column N
    const int hi   = lane >> 4;     // which 16-lane half
    const long wave = (long)(blockIdx.x) * (blockDim.x >> 5) + (threadIdx.x >> 5);
    const int base = (int)(wave * 16);
    if (base >= B) return;          // wave-uniform
    const int bidx = min(base + m, B - 1);

    // ---- Build constant A tiles (f16, 16x32 each); tile t = gate type t ----
    // A layout used here (contiguous K, B matches): lane M = lane&15 ; reg r holds K pair {K0,K0+1} with
    // K0 = 2r + 16*hi. Compute both halves with constant K, then
    // one v_cndmask per packed reg. Row m==15 is zero pad (clamped row + mask).
    const float vmask = (m == 15) ? 0.0f : 1.0f;
    const int   row_m = (m == 15) ? 14 : m;

    v16h A0[4], A1[4];
#pragma unroll
    for (int t = 0; t < 4; ++t) {
        const int row = 15 * t + row_m;
        vi8 a0i, a1i;
#pragma unroll
        for (int r = 0; r < 8; ++r) {
            const int Klo = 2 * r;  // low-half K pair base
            const int Khi = Klo + 16;                     // high-half K pair base
            const int l0 = pk2(vmask * a0e(Klo,     row, Whh0, Wih0, bi0, bh0),
                               vmask * a0e(Klo + 1, row, Whh0, Wih0, bi0, bh0));
            const int h0 = pk2(vmask * a0e(Khi,     row, Whh0, Wih0, bi0, bh0),
                               vmask * a0e(Khi + 1, row, Whh0, Wih0, bi0, bh0));
            const int l1 = pk2(vmask * a1e(Klo,     row, Wih1, Whh1, bi1, bh1),
                               vmask * a1e(Klo + 1, row, Wih1, Whh1, bi1, bh1));
            const int h1v = pk2(vmask * a1e(Khi,     row, Wih1, Whh1, bi1, bh1),
                                vmask * a1e(Khi + 1, row, Wih1, Whh1, bi1, bh1));
            a0i[r] = hi ? h0 : l0;
            a1i[r] = hi ? h1v : l1;
        }
        A0[t] = __builtin_bit_cast(v16h, a0i);
        A1[t] = __builtin_bit_cast(v16h, a1i);
    }

    // States: reg r holds hidden unit j = r + 8*hi for batch sample (base+m).
    float h1[8], c1[8], h2[8], c2[8];
#pragma unroll
    for (int r = 0; r < 8; ++r) { h1[r] = 0.f; c1[r] = 0.f; h2[r] = 0.f; c2[r] = 0.f; }

    const float beff = head[15];
    float acc = 0.0f;
    const v8f Z = {};

#pragma unroll 1
    for (int t = 0; t < 15; ++t) {
        const float xt = x[(long)bidx * 15 + t];
        const float wt = head[t];

        // ================= layer 0 =================
        // B (K=32 x N=16): low lanes hold K0..15 = [h1(15) | 1.0],
        // high lanes hold K16..31 = [x_t, 0...]. Lane column N = batch.
        float p1[8];
#pragma unroll
        for (int r = 0; r < 8; ++r) p1[r] = lane_xor16(h1[r]);

        vi8 bi;
        {
            const int xpk = pk2(xt, 0.0f);
#pragma unroll
            for (int r = 0; r < 8; ++r) {
                const int e0 = 2 * r, e1 = 2 * r + 1;
                const float l0 = (e0 < 8) ? h1[e0] : ((e0 < 15) ? p1[e0 - 8] : 1.0f);
                const float l1 = (e1 < 8) ? h1[e1] : ((e1 < 15) ? p1[e1 - 8] : 1.0f);
                const int lo = pk2(l0, l1);
                const int hv = (r == 0) ? xpk : 0;
                bi[r] = hi ? hv : lo;
            }
        }
        v16h Bv = __builtin_bit_cast(v16h, bi);

        v8f Gi = __builtin_amdgcn_wmma_f32_16x16x32_f16(false, A0[0], false, Bv, (short)0, Z, false, false);
        v8f Gf = __builtin_amdgcn_wmma_f32_16x16x32_f16(false, A0[1], false, Bv, (short)0, Z, false, false);
        v8f Gg = __builtin_amdgcn_wmma_f32_16x16x32_f16(false, A0[2], false, Bv, (short)0, Z, false, false);
        v8f Go = __builtin_amdgcn_wmma_f32_16x16x32_f16(false, A0[3], false, Bv, (short)0, Z, false, false);

#pragma unroll
        for (int r = 0; r < 8; ++r) {
            const float ig = fsig(Gi[r]);
            const float fg = fsig(Gf[r]);
            const float gg = ftanh(Gg[r]);
            const float og = fsig(Go[r]);
            const float c  = fg * c1[r] + ig * gg;
            c1[r] = c;
            h1[r] = og * ftanh(c);
        }

        // ================= layer 1 =================
        // B: low lanes K0..15 = [h1_new(15) | 1.0], high lanes K16..31 = [h2(15) | 0]
        float q1[8], q2[8];
#pragma unroll
        for (int r = 0; r < 8; ++r) { q1[r] = lane_xor16(h1[r]); q2[r] = lane_xor16(h2[r]); }

#pragma unroll
        for (int r = 0; r < 8; ++r) {
            const int e0 = 2 * r, e1 = 2 * r + 1;
            const float l0 = (e0 < 8) ? h1[e0] : ((e0 < 15) ? q1[e0 - 8] : 1.0f);
            const float l1 = (e1 < 8) ? h1[e1] : ((e1 < 15) ? q1[e1 - 8] : 1.0f);
            const float g0 = (e0 < 8) ? q2[e0] : ((e0 < 15) ? h2[e0 - 8] : 0.0f);
            const float g1 = (e1 < 8) ? q2[e1] : ((e1 < 15) ? h2[e1 - 8] : 0.0f);
            bi[r] = hi ? pk2(g0, g1) : pk2(l0, l1);
        }
        Bv = __builtin_bit_cast(v16h, bi);

        Gi = __builtin_amdgcn_wmma_f32_16x16x32_f16(false, A1[0], false, Bv, (short)0, Z, false, false);
        Gf = __builtin_amdgcn_wmma_f32_16x16x32_f16(false, A1[1], false, Bv, (short)0, Z, false, false);
        Gg = __builtin_amdgcn_wmma_f32_16x16x32_f16(false, A1[2], false, Bv, (short)0, Z, false, false);
        Go = __builtin_amdgcn_wmma_f32_16x16x32_f16(false, A1[3], false, Bv, (short)0, Z, false, false);

#pragma unroll
        for (int r = 0; r < 8; ++r) {
            const float ig = fsig(Gi[r]);
            const float fg = fsig(Gf[r]);
            const float gg = ftanh(Gg[r]);
            const float og = fsig(Go[r]);
            const float c  = fg * c2[r] + ig * gg;
            c2[r] = c;
            h2[r] = og * ftanh(c);
        }

        // feat[:, t] = h2 hidden unit 14 = slot r=6 in the high half.
        acc = __builtin_fmaf(wt, h2[6], acc);
    }

    // High lane L+16 owns batch column L's reduced output.
    if (hi && (base + m) < B) out[base + m] = acc + beff;
}

extern "C" void kernel_launch(void* const* d_in, const int* in_sizes, int n_in,
                              void* d_out, int out_size, void* d_ws, size_t ws_size,
                              hipStream_t stream) {
    const float* x    = (const float*)d_in[0];
    const float* Wih0 = (const float*)d_in[1];
    const float* Whh0 = (const float*)d_in[2];
    const float* bi0  = (const float*)d_in[3];
    const float* bh0  = (const float*)d_in[4];
    const float* Wih1 = (const float*)d_in[5];
    const float* Whh1 = (const float*)d_in[6];
    const float* bi1  = (const float*)d_in[7];
    const float* bh1  = (const float*)d_in[8];
    const float* W1   = (const float*)d_in[9];
    const float* b1   = (const float*)d_in[10];
    const float* W2   = (const float*)d_in[11];
    const float* b2   = (const float*)d_in[12];
    const float* W3   = (const float*)d_in[13];
    const float* b3   = (const float*)d_in[14];

    const int B = in_sizes[0] / 15;           // x is (B, 15, 1)
    float* head = (float*)d_ws;               // 16 floats: w_eff[15], b_eff

    head_collapse_kernel<<<1, 32, 0, stream>>>(W1, b1, W2, b2, W3, b3, head);

    const int waves   = (B + 15) / 16;        // one wave32 per 16 samples
    const int threads = 256;                  // 8 waves per block
    const int blocks  = (waves * 32 + threads - 1) / threads;
    lstm_fused_wmma<<<blocks, threads, 0, stream>>>(
        x, Wih0, Whh0, bi0, bh0, Wih1, Whh1, bi1, bh1,
        head, (float*)d_out, B);
}
